// PerLayerLoRAHyperNetwork_57878979281031
// MI455X (gfx1250) — compile-verified, matrix operand order checked by reading, not run on hardware
//
#include <hip/hip_runtime.h>

// ---------------------------------------------------------------------------
// PerLayerLoRAHyperNetwork for MI455X (gfx1250, wave32, WMMA).
// bf16 WMMA (v_wmma_f32_16x16x32_bf16) for all GEMMs; VALU for the
// store-bandwidth-bound rank-8 outer-product expansion.
// ---------------------------------------------------------------------------

#define B_TOT   256
#define T_WIN   64
#define N_UIN   16
#define HID     512
#define G4H     2048        // 4*HID
#define KC      544         // 512 (h) + 16 (u_t) + 16 zero pad = 17 * 32
#define NKC     17
#define MLPD    1024
#define EMBD    64
#define INM     576         // HID + EMBD = 18 * 32
#define RANK    8
#define RAWSTR  4096        // max rank*(din+dout), per-layer raw stride
#define ENCCOLS 140288      // per-batch columns of delta_enc (== delta_dec)
#define LSTM_T  512         // 16 waves

typedef __attribute__((ext_vector_type(16))) __bf16 bf16x16;
typedef __attribute__((ext_vector_type(8)))  float  floatx8;

// Scheduler shaping: per k-chunk, group {DS reads, VMEM reads, WMMAs} so all
// fragment loads of a chunk issue before its WMMAs (one wait per chunk).
#if defined(__has_builtin)
#if __has_builtin(__builtin_amdgcn_sched_group_barrier)
#define SCHED_CHUNK(nds, nvm, nmm)                         \
  do {                                                     \
    __builtin_amdgcn_sched_group_barrier(0x100, nds, 0);   \
    __builtin_amdgcn_sched_group_barrier(0x020, nvm, 0);   \
    __builtin_amdgcn_sched_group_barrier(0x008, nmm, 0);   \
  } while (0)
#endif
#endif
#ifndef SCHED_CHUNK
#define SCHED_CHUNK(nds, nvm, nmm) do {} while (0)
#endif

// ---- helpers --------------------------------------------------------------

__device__ __forceinline__ unsigned short f2bf(float f) {
  union { float f; unsigned u; } v; v.f = f;
  unsigned r = v.u + 0x7FFFu + ((v.u >> 16) & 1u);   // round-to-nearest-even
  return (unsigned short)(r >> 16);
}

// 16 bf16 elements (32B) -> A/B fragment, as two b128 loads (LDS or global)
__device__ __forceinline__ bf16x16 frag_ld(const unsigned short* p) {
  union { uint4 q[2]; bf16x16 v; } u;
  u.q[0] = *reinterpret_cast<const uint4*>(p);
  u.q[1] = *reinterpret_cast<const uint4*>(p + 8);
  return u.v;
}

// 16 f32 -> bf16 fragment (on-the-fly conversion for head weights)
__device__ __forceinline__ bf16x16 frag_cvt(const float* p) {
  union { unsigned short e[16]; bf16x16 v; } u;
#pragma unroll
  for (int i = 0; i < 16; ++i) u.e[i] = f2bf(p[i]);
  return u.v;
}

__device__ __forceinline__ floatx8 zero8() {
  floatx8 z = {0.f, 0.f, 0.f, 0.f, 0.f, 0.f, 0.f, 0.f};
  return z;
}

__device__ __forceinline__ floatx8 wmma_bf16(bf16x16 a, bf16x16 b, floatx8 c) {
  // (neg_a, A, neg_b, B, c_mod, C, reuse_a, reuse_b)
  return __builtin_amdgcn_wmma_f32_16x16x32_bf16(false, a, false, b,
                                                 (short)0, c, false, false);
}

__device__ __forceinline__ float sigmoidf_(float x) {
  return 1.0f / (1.0f + __expf(-x));
}
__device__ __forceinline__ float tanhf_fast(float x) {
  return 2.0f / (1.0f + __expf(-2.0f * x)) - 1.0f;
}

// ---- kernel 1: weight prep (fused [Whh|Wih|pad] -> bf16, biases, Wb1/2) ----

__global__ __launch_bounds__(256) void prep_kernel(
    const float* __restrict__ Wih, const float* __restrict__ Whh,
    const float* __restrict__ bih, const float* __restrict__ bhh,
    const float* __restrict__ Wb1, const float* __restrict__ Wb2,
    unsigned short* __restrict__ Wc, float* __restrict__ bias_c,
    unsigned short* __restrict__ Wb1b, unsigned short* __restrict__ Wb2b) {
  const int t0 = blockIdx.x * 256 + threadIdx.x;
  const int stride = gridDim.x * 256;
  for (int i = t0; i < G4H * KC; i += stride) {
    int g = i / KC, k = i - g * KC;
    float v = (k < HID) ? Whh[g * HID + k]
            : (k < HID + N_UIN ? Wih[g * N_UIN + (k - HID)] : 0.0f);
    Wc[i] = f2bf(v);
  }
  for (int i = t0; i < G4H; i += stride) bias_c[i] = bih[i] + bhh[i];
  for (int i = t0; i < MLPD * INM; i += stride) Wb1b[i] = f2bf(Wb1[i]);
  for (int i = t0; i < MLPD * MLPD; i += stride) Wb2b[i] = f2bf(Wb2[i]);
}

// ---- kernel 2: per-batch input RMS ----------------------------------------

__global__ __launch_bounds__(256) void rms_kernel(const float* __restrict__ u,
                                                  float* __restrict__ urms) {
  __shared__ float red[256];
  const int b = blockIdx.x, tid = threadIdx.x;
  float s = 0.0f;
  for (int i = tid; i < T_WIN * N_UIN; i += 256) {
    float v = u[(size_t)b * (T_WIN * N_UIN) + i];
    s += v * v;
  }
  red[tid] = s;
  __syncthreads();
  for (int off = 128; off > 0; off >>= 1) {
    if (tid < off) red[tid] += red[tid + off];
    __syncthreads();
  }
  if (tid == 0) urms[b] = __fsqrt_rn(red[0] * (1.0f / (T_WIN * N_UIN)));
}

// ---- kernel 3: LSTM recurrence (h in LDS bf16, c in accum-layout VGPRs) ----
// 16 workgroups x 16 batch rows, 16 waves each. Wave w owns h-columns
// [32w, 32w+32): per step, 4 gates x 2 n-tiles = 8 WMMA tiles (j = g*2+n4)
// over K = 544 = [h | u_t | pad]; then the wave-local c/h update.
// sched_group_barrier enforces {2 ds, 16 vmem, 8 wmma} per k-chunk so the
// L2 round trip is paid once per chunk, overlapped with the previous
// chunk's WMMAs, with 4 waves/SIMD on top.

__global__ __launch_bounds__(LSTM_T, 1) void lstm_kernel(
    const float* __restrict__ u, const unsigned short* __restrict__ Wc,
    const float* __restrict__ bias_c, unsigned short* __restrict__ h_out) {
  __shared__ unsigned short in_sh[16][KC];   // 17,408 B
  const int btile = blockIdx.x;
  const int tid = threadIdx.x;
  const int wave = tid >> 5, lane = tid & 31;
  const int lrow = lane & 15, lhi = lane >> 4;
  const int hc0 = wave * 32;                 // 16 waves x 32 h-cols

  for (int i = tid; i < 16 * KC; i += LSTM_T) (&in_sh[0][0])[i] = 0;

  floatx8 cst[2];
#pragma unroll
  for (int n = 0; n < 2; ++n) cst[n] = zero8();

  float bI[2], bF[2], bG[2], bO[2];
#pragma unroll
  for (int n4 = 0; n4 < 2; ++n4) {
    int col = hc0 + n4 * 16 + lrow;
    bI[n4] = bias_c[col];
    bF[n4] = bias_c[HID + col];
    bG[n4] = bias_c[2 * HID + col];
    bO[n4] = bias_c[3 * HID + col];
  }

  // Per-lane base element offset into Wc (row = hc0+lrow, k = lhi*16).
  const unsigned wbase = (unsigned)(hc0 + lrow) * KC + (unsigned)lhi * 16;

#pragma clang loop unroll(disable)
  for (int t = 0; t < T_WIN; ++t) {
    // Loop-variant (opaque) zero offset: blocks hoisting of the weight
    // fragment loads across steps without losing global provenance.
    unsigned long zoff = 0;
    asm volatile("" : "+s"(zoff));
    const unsigned short* Wct = Wc + zoff;

    if (tid < 256) {  // stage u_t (16 rows x 16) into the K=[512,528) slot
      int r = tid >> 4, c = tid & 15;
      in_sh[r][HID + c] =
          f2bf(u[((size_t)(btile * 16 + r) * T_WIN + t) * N_UIN + c]);
    }
    __syncthreads();  // h (prev step) + u_t visible

    floatx8 acc[8];   // j = gate*2 + n4
#pragma unroll
    for (int j = 0; j < 8; ++j) acc[j] = zero8();

#pragma unroll
    for (int kc = 0; kc < NKC; ++kc) {
      bf16x16 a = frag_ld(&in_sh[lrow][kc * 32 + lhi * 16]);
      bf16x16 bfr[8];
#pragma unroll
      for (int j = 0; j < 8; ++j) {           // batch: 16 b128 loads in flight
        // tile offset is a compile-time constant -> instruction ioffset
        const unsigned toff = (unsigned)((j >> 1) * HID + (j & 1) * 16) * KC;
        bfr[j] = frag_ld(Wct + wbase + toff + kc * 32);
      }
#pragma unroll
      for (int j = 0; j < 8; ++j) acc[j] = wmma_bf16(a, bfr[j], acc[j]);
      SCHED_CHUNK(2, 16, 8);
    }
    __syncthreads();  // all reads of in_sh done before rewriting h

#pragma unroll
    for (int n4 = 0; n4 < 2; ++n4) {
      int col = hc0 + n4 * 16 + lrow;
#pragma unroll
      for (int r = 0; r < 8; ++r) {
        float iv = sigmoidf_(acc[n4][r] + bI[n4]);
        float fv = sigmoidf_(acc[2 + n4][r] + bF[n4]);
        float gv = tanhf_fast(acc[4 + n4][r] + bG[n4]);
        float ov = sigmoidf_(acc[6 + n4][r] + bO[n4]);
        float cn = fv * cst[n4][r] + iv * gv;
        cst[n4][r] = cn;
        in_sh[r + 8 * lhi][col] = f2bf(ov * tanhf_fast(cn));  // h in D layout
      }
    }
  }
  __syncthreads();
  for (int i = tid; i < 16 * HID; i += LSTM_T) {
    int r = i >> 9, c = i & (HID - 1);
    h_out[(size_t)(btile * 16 + r) * HID + c] = in_sh[r][c];
  }
}

// ---- kernel 4: conditioned MLP feat = tanh(tanh([h|e]W1^T+b1)W2^T+b2) -----

__global__ __launch_bounds__(256, 2) void mlp_kernel(
    const unsigned short* __restrict__ h_bf, const float* __restrict__ emb,
    const unsigned short* __restrict__ Wb1b, const float* __restrict__ bb1,
    const unsigned short* __restrict__ Wb2b, const float* __restrict__ bb2,
    unsigned short* __restrict__ feat_all) {
  __shared__ unsigned short in_sh[16][INM];    // 18,432 B
  __shared__ unsigned short f1_sh[16][MLPD];   // 32,768 B
  const int btile = blockIdx.x, layer = blockIdx.y;
  const int tid = threadIdx.x;
  const int wave = tid >> 5, lane = tid & 31;
  const int lrow = lane & 15, lhi = lane >> 4;

  for (int i = tid; i < 16 * HID; i += 256) {
    int r = i >> 9, c = i & (HID - 1);
    in_sh[r][c] = h_bf[(size_t)(btile * 16 + r) * HID + c];
  }
  for (int i = tid; i < 16 * EMBD; i += 256) {
    int r = i >> 6, c = i & (EMBD - 1);
    in_sh[r][HID + c] = f2bf(emb[layer * EMBD + c]);
  }
  __syncthreads();

  floatx8 acc[8];
#pragma unroll
  for (int t2 = 0; t2 < 8; ++t2) acc[t2] = zero8();
  {
    const unsigned wb = (unsigned)(wave * 128 + lrow) * INM + (unsigned)lhi * 16;
#pragma unroll
    for (int kc = 0; kc < INM / 32; ++kc) {
      bf16x16 a = frag_ld(&in_sh[lrow][kc * 32 + lhi * 16]);
      bf16x16 bfr[8];
#pragma unroll
      for (int t2 = 0; t2 < 8; ++t2)
        bfr[t2] = frag_ld(Wb1b + wb + (unsigned)(t2 * 16) * INM + kc * 32);
#pragma unroll
      for (int t2 = 0; t2 < 8; ++t2) acc[t2] = wmma_bf16(a, bfr[t2], acc[t2]);
      SCHED_CHUNK(2, 16, 8);
    }
  }
#pragma unroll
  for (int t2 = 0; t2 < 8; ++t2) {
    int col = wave * 128 + t2 * 16 + lrow;
    float bias = bb1[col];
#pragma unroll
    for (int r = 0; r < 8; ++r)
      f1_sh[r + 8 * lhi][col] = f2bf(tanhf_fast(acc[t2][r] + bias));
  }
  __syncthreads();

#pragma unroll
  for (int t2 = 0; t2 < 8; ++t2) acc[t2] = zero8();
  {
    const unsigned wb = (unsigned)(wave * 128 + lrow) * MLPD + (unsigned)lhi * 16;
#pragma unroll
    for (int kc = 0; kc < MLPD / 32; ++kc) {
      bf16x16 a = frag_ld(&f1_sh[lrow][kc * 32 + lhi * 16]);
      bf16x16 bfr[8];
#pragma unroll
      for (int t2 = 0; t2 < 8; ++t2)
        bfr[t2] = frag_ld(Wb2b + wb + (unsigned)(t2 * 16) * MLPD + kc * 32);
#pragma unroll
      for (int t2 = 0; t2 < 8; ++t2) acc[t2] = wmma_bf16(a, bfr[t2], acc[t2]);
      SCHED_CHUNK(2, 16, 8);
    }
  }
#pragma unroll
  for (int t2 = 0; t2 < 8; ++t2) {
    int col = wave * 128 + t2 * 16 + lrow;
    float bias = bb2[col];
#pragma unroll
    for (int r = 0; r < 8; ++r)
      feat_all[((size_t)layer * B_TOT + btile * 16 + r + 8 * lhi) * MLPD + col] =
          f2bf(tanhf_fast(acc[t2][r] + bias));
  }
}

// ---- kernel 5: per-layer head GEMM raw = feat @ head^T --------------------
// 2048 blocks -> latency hidden by occupancy; keep simple guarded structure.

struct HeadPtrs { const float* w[8]; };

__global__ __launch_bounds__(256) void head_kernel(
    const unsigned short* __restrict__ feat_all, HeadPtrs hp,
    float* __restrict__ raw_all) {
  __shared__ unsigned short f_sh[16][MLPD];  // 32,768 B
  const int DIN[8]  = {16, 256, 256, 256, 20, 256, 256, 256};
  const int DOUT[8] = {256, 256, 256, 20, 256, 256, 256, 16};
  const int btile = blockIdx.x, ngrp = blockIdx.y, layer = blockIdx.z;
  const int NR = RANK * (DIN[layer] + DOUT[layer]);   // multiple of 16
  const int nbase = ngrp * 256;
  if (nbase >= NR) return;  // block-uniform
  const int tid = threadIdx.x;
  const int wave = tid >> 5, lane = tid & 31;
  const int lrow = lane & 15, lhi = lane >> 4;

  for (int i = tid; i < 16 * MLPD; i += 256) {
    int r = i >> 10, c = i & (MLPD - 1);
    f_sh[r][c] = feat_all[((size_t)layer * B_TOT + btile * 16 + r) * MLPD + c];
  }
  __syncthreads();

  const float* W = hp.w[layer];
  floatx8 acc[2];
  acc[0] = zero8(); acc[1] = zero8();
  const int c0 = nbase + wave * 32;
  const bool v0 = (c0 < NR), v1 = (c0 + 16 < NR);  // wave-uniform guards
  for (int kc = 0; kc < MLPD / 32; ++kc) {
    bf16x16 a = frag_ld(&f_sh[lrow][kc * 32 + lhi * 16]);
    if (v0) {
      bf16x16 b = frag_cvt(&W[(size_t)(c0 + lrow) * MLPD + kc * 32 + lhi * 16]);
      acc[0] = wmma_bf16(a, b, acc[0]);
    }
    if (v1) {
      bf16x16 b =
          frag_cvt(&W[(size_t)(c0 + 16 + lrow) * MLPD + kc * 32 + lhi * 16]);
      acc[1] = wmma_bf16(a, b, acc[1]);
    }
  }
#pragma unroll
  for (int t2 = 0; t2 < 2; ++t2) {
    if ((t2 == 0 && v0) || (t2 == 1 && v1)) {
      int col = c0 + t2 * 16 + lrow;
#pragma unroll
      for (int r = 0; r < 8; ++r)
        raw_all[((size_t)layer * B_TOT + btile * 16 + r + 8 * lhi) * RAWSTR +
                col] = acc[t2][r];
    }
  }
}

// ---- kernel 6: rank-8 outer product + scale*u_rms (store-BW bound) --------

__global__ __launch_bounds__(256) void outer_kernel(
    const float* __restrict__ raw, const float* __restrict__ urms,
    const float* __restrict__ scale_p, float* __restrict__ out,
    int din, int dout, int loff, long total) {
  long idx = ((long)blockIdx.x * 256 + threadIdx.x) * 4;
  if (idx >= total) return;
  const int dd = din * dout;            // multiple of 4
  const int b = (int)(idx / dd);
  const int rem = (int)(idx - (long)b * dd);
  const int i = rem / dout;
  const int o = rem - i * dout;         // multiple of 4 (dout % 4 == 0)
  const float* rb = raw + (size_t)b * RAWSTR;
  float av[RANK];
#pragma unroll
  for (int r = 0; r < RANK; ++r) av[r] = rb[i * RANK + r];
  const float* bm = rb + din * RANK;
  float s0 = 0.f, s1 = 0.f, s2 = 0.f, s3 = 0.f;
#pragma unroll
  for (int r = 0; r < RANK; ++r) {
    const float* bp = bm + r * dout + o;
    s0 += av[r] * bp[0]; s1 += av[r] * bp[1];
    s2 += av[r] * bp[2]; s3 += av[r] * bp[3];
  }
  const float sc = scale_p[0] * urms[b];
  float* op = out + (size_t)b * ENCCOLS + loff + (size_t)i * dout + o;
  op[0] = s0 * sc; op[1] = s1 * sc; op[2] = s2 * sc; op[3] = s3 * sc;
}

// ---- host launch ----------------------------------------------------------
// Workspace layout (bytes, 256-aligned; ~43.5 MB total):
//   [0)        Wc bf16        2048*544*2  = 2,228,224
//   [2228224)  bias_c f32     2048*4      = 8,192
//   [2236416)  Wb1 bf16       1024*576*2  = 1,179,648
//   [3416064)  Wb2 bf16       1024*1024*2 = 2,097,152
//   [5513216)  h bf16         256*512*2   = 262,144
//   [5775360)  u_rms f32      256*4       = 1,024
//   [5776384)  feat bf16      8*256*1024*2= 4,194,304
//   [9970688)  raw f32        8*256*4096*4= 33,554,432

extern "C" void kernel_launch(void* const* d_in, const int* in_sizes, int n_in,
                              void* d_out, int out_size, void* d_ws,
                              size_t ws_size, hipStream_t stream) {
  const float* u    = (const float*)d_in[0];
  const float* Wih  = (const float*)d_in[1];
  const float* Whh  = (const float*)d_in[2];
  const float* bih  = (const float*)d_in[3];
  const float* bhh  = (const float*)d_in[4];
  const float* emb  = (const float*)d_in[5];
  const float* Wb1  = (const float*)d_in[6];
  const float* bb1  = (const float*)d_in[7];
  const float* Wb2  = (const float*)d_in[8];
  const float* bb2  = (const float*)d_in[9];
  HeadPtrs hp;
  for (int l = 0; l < 8; ++l) hp.w[l] = (const float*)d_in[10 + l];
  const float* scale_p = (const float*)d_in[18];
  float* out = (float*)d_out;

  char* ws = (char*)d_ws;
  unsigned short* Wc     = (unsigned short*)(ws + 0);
  float*          bias_c = (float*)(ws + 2228224);
  unsigned short* Wb1b   = (unsigned short*)(ws + 2236416);
  unsigned short* Wb2b   = (unsigned short*)(ws + 3416064);
  unsigned short* h_bf   = (unsigned short*)(ws + 5513216);
  float*          urms   = (float*)(ws + 5775360);
  unsigned short* featA  = (unsigned short*)(ws + 5776384);
  float*          rawA   = (float*)(ws + 9970688);

  prep_kernel<<<512, 256, 0, stream>>>(Wih, Whh, bih, bhh, Wb1, Wb2,
                                       Wc, bias_c, Wb1b, Wb2b);
  rms_kernel<<<B_TOT, 256, 0, stream>>>(u, urms);
  lstm_kernel<<<B_TOT / 16, LSTM_T, 0, stream>>>(u, Wc, bias_c, h_bf);
  mlp_kernel<<<dim3(B_TOT / 16, 8), 256, 0, stream>>>(h_bf, emb, Wb1b, bb1,
                                                      Wb2b, bb2, featA);
  head_kernel<<<dim3(B_TOT / 16, 16, 8), 256, 0, stream>>>(featA, hp, rawA);

  static const int DIN[8]  = {16, 256, 256, 256, 20, 256, 256, 256};
  static const int DOUT[8] = {256, 256, 256, 20, 256, 256, 256, 16};
  static const int LOFF[8] = {0, 4096, 69632, 135168, 0, 5120, 70656, 136192};
  for (int l = 0; l < 8; ++l) {
    long total = (long)B_TOT * DIN[l] * DOUT[l];
    int blocks = (int)((total / 4 + 255) / 256);
    float* outH = out + (l < 4 ? (size_t)0 : (size_t)B_TOT * ENCCOLS);
    outer_kernel<<<blocks, 256, 0, stream>>>(
        rawA + (size_t)l * B_TOT * RAWSTR, urms, scale_p, outH,
        DIN[l], DOUT[l], LOFF[l], total);
  }
}
